// Attention_1529008357382
// MI455X (gfx1250) — matrix operand order repaired and checked by reading, not compiled, not run on hardware
//
#include <hip/hip_runtime.h>

// ---- problem constants ----
#define B_     8
#define N_     4096
#define C_     256
#define HEADS_ 8
#define HD_    32
#define NK_    256      // (64/4)^2

typedef unsigned int u32;
typedef unsigned short u16;
typedef __attribute__((ext_vector_type(16))) _Float16 v16h;
typedef __attribute__((ext_vector_type(8)))  float    v8f;
typedef __attribute__((ext_vector_type(4)))  u32      u32x4;
typedef __attribute__((ext_vector_type(2)))  u32      u32x2;
typedef __attribute__((ext_vector_type(4)))  float    f32x4;

union Frag16 { u32x4 q[2]; v16h v; };        // 32B: one 16-element f16 fragment
union H8     { u32x4 q;    _Float16 h[8]; }; // 16B
union H4     { u32x2 d;    _Float16 h[4]; }; // 8B

__device__ __forceinline__ v8f wmma_f16(v16h a, v16h b, v8f c) {
  return __builtin_amdgcn_wmma_f32_16x16x32_f16(false, a, false, b, (short)0, c,
                                                false, false);
}

__device__ __forceinline__ v8f zero_v8f() {
  v8f z;
#pragma unroll
  for (int i = 0; i < 8; ++i) z[i] = 0.0f;
  return z;
}

// ---- gfx1250 async global->LDS copy (ASYNCcnt) -------------------------------
__device__ __forceinline__ void async_ld_b128(const void* gptr, void* lptr) {
  u32 lds = (u32)(uintptr_t)lptr;   // low 32 bits of generic LDS ptr = LDS addr
  asm volatile("global_load_async_to_lds_b128 %0, %1, off"
               :: "v"(lds), "v"(gptr) : "memory");
}

__device__ __forceinline__ void wait_async0() {
#if __has_builtin(__builtin_amdgcn_s_wait_asynccnt)
  __builtin_amdgcn_s_wait_asynccnt(0);
#else
  asm volatile("s_wait_asynccnt 0x0" ::: "memory");
#endif
}

// A fragment (16x32 f16). p -> element (row0,k0); row stride ld (multiple of 8).
__device__ __forceinline__ v16h frag_a(const _Float16* p, int ld) {
  const int lane = threadIdx.x & 31;
  const int row  = lane & 15;
  const int kh   = (lane >> 4) << 3;   // 0 or 8
  Frag16 f;
  f.q[0] = *(const u32x4*)(p + row * ld + kh);
  f.q[1] = *(const u32x4*)(p + row * ld + 16 + kh);
  return f.v;
}

// B fragment (32x16 f16) from a transposed [n][k] tile; n stride ld (mult of 8).
__device__ __forceinline__ v16h frag_b(const _Float16* p, int ld) {
  const int lane = threadIdx.x & 31;
  const int n    = lane & 15;
  const int kh   = (lane >> 4) << 3;   // 0 or 8
  Frag16 f;
  f.q[0] = *(const u32x4*)(p + n * ld + kh);
  f.q[1] = *(const u32x4*)(p + n * ld + 16 + kh);
  return f.v;
}

// ------------- weight packing: f32 -> f16, TRANSPOSED to [n][k] ---------------
__global__ void __launch_bounds__(256) k_pack_w(
    const float* __restrict__ qw, const float* __restrict__ kvw,
    const float* __restrict__ pw, const float* __restrict__ srw,
    _Float16* __restrict__ qwT, _Float16* __restrict__ kvwT,
    _Float16* __restrict__ pwT, _Float16* __restrict__ srwT) {
  int i = blockIdx.x * 256 + threadIdx.x;            // up to 4096*256
  if (i < C_ * C_) {                                 // i = n*256 + k
    int n = i >> 8, k = i & 255;
    qwT[i] = (_Float16)qw[k * C_ + n];
    pwT[i] = (_Float16)pw[k * C_ + n];
  }
  if (i < 2 * C_ * C_) {                             // i = n*256 + k, n<512
    int n = i >> 8, k = i & 255;
    kvwT[i] = (_Float16)kvw[k * 2 * C_ + n];
  }
  // srw OIHW [co][ci][ky][kx] -> srwT[co][kidx], kidx=(ky*4+kx)*256+ci
  {
    int co = i >> 12, kidx = i & 4095;
    int ci = kidx & 255, p = kidx >> 8;
    srwT[i] = (_Float16)srw[co * 4096 + ci * 16 + p];
  }
}

// ---------------- x: f32 -> f16 ----------------
__global__ void __launch_bounds__(256) k_cast_x(const float* __restrict__ x,
                                                _Float16* __restrict__ xh) {
  int i = (blockIdx.x * 256 + threadIdx.x) * 4;
  f32x4 f = *(const f32x4*)(x + i);
  H4 u;
#pragma unroll
  for (int j = 0; j < 4; ++j) u.h[j] = (_Float16)f[j];
  *(u32x2*)(xh + i) = u.d;
}

// ------- q = x @ q_w (M=32768,N=256,K=256), 256x64 tile, 2 row-tiles/wave -----
__global__ void __launch_bounds__(256) k_gemm_q(const _Float16* __restrict__ xh,
                                                const _Float16* __restrict__ qwT,
                                                _Float16* __restrict__ qh) {
  __shared__ _Float16 sA[2][256 * 40];
  __shared__ _Float16 sB[2][64 * 40];
  const int bm = blockIdx.x;              // rows bm*256
  const int colBase = blockIdx.y * 64;
  const int t = threadIdx.x, wv = t >> 5;
  const int r4 = t >> 2, c8 = (t & 3) * 8;     // 64 rows x 4 chunks
  v8f acc0[4], acc1[4];
#pragma unroll
  for (int c = 0; c < 4; ++c) { acc0[c] = zero_v8f(); acc1[c] = zero_v8f(); }

  auto issue = [&](int buf, int k0) {
#pragma unroll
    for (int j = 0; j < 4; ++j)
      async_ld_b128(xh + (size_t)(bm * 256 + j * 64 + r4) * C_ + k0 + c8,
                    &sA[buf][(j * 64 + r4) * 40 + c8]);
    async_ld_b128(qwT + (size_t)(colBase + r4) * C_ + k0 + c8,
                  &sB[buf][r4 * 40 + c8]);
  };
  issue(0, 0);
#pragma unroll
  for (int s = 0; s < 8; ++s) {
    int buf = s & 1;
    wait_async0();
    __syncthreads();
    if (s + 1 < 8) issue(buf ^ 1, (s + 1) * 32);
    v16h a0 = frag_a(sA[buf] + (wv * 16) * 40, 40);
    v16h a1 = frag_a(sA[buf] + ((128 + wv * 16)) * 40, 40);
#pragma unroll
    for (int ct = 0; ct < 4; ++ct) {
      v16h bf = frag_b(sB[buf] + ct * 16 * 40, 40);
      acc0[ct] = wmma_f16(a0, bf, acc0[ct]);
      acc1[ct] = wmma_f16(a1, bf, acc1[ct]);
    }
  }
  const int lane = t & 31;
  const int n = lane & 15, mh = (lane >> 4) << 3;
  const float scale = 0.17677669529663687f;   // hd^-0.5
#pragma unroll
  for (int half = 0; half < 2; ++half) {
    v8f* acc = half ? acc1 : acc0;
#pragma unroll
    for (int ct = 0; ct < 4; ++ct) {
      int c = colBase + ct * 16 + n;
      int head = c >> 5, d = c & 31;
#pragma unroll
      for (int r = 0; r < 8; ++r) {
        int row = bm * 256 + half * 128 + wv * 16 + mh + r;
        int b = row >> 12, nn = row & 4095;
        qh[((size_t)(b * HEADS_ + head) * N_ + nn) * HD_ + d] =
            (_Float16)(acc[ct][r] * scale);
      }
    }
  }
}

// ------- SR conv as implicit GEMM (M=2048,N=256,K=4096) -> f32 + bias ---------
__global__ void __launch_bounds__(256) k_conv_gemm(const float* __restrict__ x,
                                                   const _Float16* __restrict__ srwT,
                                                   const float* __restrict__ srb,
                                                   float* __restrict__ xr) {
  __shared__ _Float16 sA[2][64 * 40];
  __shared__ _Float16 sB[2][64 * 40];
  const int bm = blockIdx.x;              // 0..31 -> rows bm*64
  const int colBase = blockIdx.y * 64;
  const int t = threadIdx.x, wv = t >> 5;
  const int rt = wv >> 1;                 // 0..3
  const int ct0 = (wv & 1) * 2;           // 0 or 2
  v8f acc[2];
  acc[0] = zero_v8f(); acc[1] = zero_v8f();
  const int arow = t >> 2, ac8 = (t & 3) * 8;
  const int grow = bm * 64 + arow;
  const int gb = grow >> 8, gnk = grow & 255;
  const int oy = gnk >> 4, ox = gnk & 15;

  auto stageA = [&](int buf, int k0) {    // gather + f32->f16 convert
    int p = k0 >> 8, ky = p >> 2, kx = p & 3, cib = k0 & 255;
    int pix = (oy * 4 + ky) * 64 + ox * 4 + kx;
    const float* src = x + ((size_t)(gb * N_ + pix)) * C_ + cib + ac8;
    f32x4 f0 = *(const f32x4*)src;
    f32x4 f1 = *(const f32x4*)(src + 4);
    H8 u;
#pragma unroll
    for (int j = 0; j < 4; ++j) { u.h[j] = (_Float16)f0[j]; u.h[4 + j] = (_Float16)f1[j]; }
    *(u32x4*)(&sA[buf][arow * 40 + ac8]) = u.q;
  };
  auto issueB = [&](int buf, int k0) {
    async_ld_b128(srwT + (size_t)(colBase + arow) * 4096 + k0 + ac8,
                  &sB[buf][arow * 40 + ac8]);
  };
  stageA(0, 0);
  issueB(0, 0);
  for (int s = 0; s < 128; ++s) {
    int buf = s & 1;
    wait_async0();
    __syncthreads();
    if (s + 1 < 128) { issueB(buf ^ 1, (s + 1) * 32); stageA(buf ^ 1, (s + 1) * 32); }
    v16h a = frag_a(sA[buf] + rt * 16 * 40, 40);
    acc[0] = wmma_f16(a, frag_b(sB[buf] + ct0 * 16 * 40, 40), acc[0]);
    acc[1] = wmma_f16(a, frag_b(sB[buf] + (ct0 + 1) * 16 * 40, 40), acc[1]);
  }
  const int lane = t & 31;
  const int n = lane & 15, mh = (lane >> 4) << 3;
#pragma unroll
  for (int cc = 0; cc < 2; ++cc) {
    int c = colBase + (ct0 + cc) * 16 + n;
    float bias = srb[c];
#pragma unroll
    for (int r = 0; r < 8; ++r) {
      int row = bm * 64 + rt * 16 + mh + r;
      xr[(size_t)row * C_ + c] = acc[cc][r] + bias;
    }
  }
}

// ---------------- LayerNorm over C, f32 -> f16 ----------------
__global__ void __launch_bounds__(256) k_ln(const float* __restrict__ xr,
                                            const float* __restrict__ lnw,
                                            const float* __restrict__ lnb,
                                            _Float16* __restrict__ xrh) {
  __shared__ float s1[8], s2[8];
  int row = blockIdx.x, t = threadIdx.x;
  float v = xr[(size_t)row * C_ + t];
  float sum = v, sq = v * v;
#pragma unroll
  for (int m = 16; m >= 1; m >>= 1) {
    sum += __shfl_xor(sum, m, 32);
    sq  += __shfl_xor(sq, m, 32);
  }
  if ((t & 31) == 0) { s1[t >> 5] = sum; s2[t >> 5] = sq; }
  __syncthreads();
  float ts = 0.f, tq = 0.f;
#pragma unroll
  for (int i = 0; i < 8; ++i) { ts += s1[i]; tq += s2[i]; }
  float mu  = ts * (1.0f / C_);
  float var = tq * (1.0f / C_) - mu * mu;
  float rs  = rsqrtf(var + 1e-5f);
  xrh[(size_t)row * C_ + t] = (_Float16)((v - mu) * rs * lnw[t] + lnb[t]);
}

// ------- kv GEMM (M=2048,N=512,K=256) -> k/v in [b][h][nk][d] -----------------
__global__ void __launch_bounds__(256) k_gemm_kv(const _Float16* __restrict__ xrh,
                                                 const _Float16* __restrict__ kvwT,
                                                 _Float16* __restrict__ kmat,
                                                 _Float16* __restrict__ vmat) {
  __shared__ _Float16 sA[2][64 * 40];
  __shared__ _Float16 sB[2][64 * 40];
  const int bm = blockIdx.x;              // 0..31
  const int colBase = blockIdx.y * 64;    // 0..7
  const int t = threadIdx.x, wv = t >> 5;
  const int rt = wv >> 1, ct0 = (wv & 1) * 2;
  const int r4 = t >> 2, c8 = (t & 3) * 8;
  v8f acc[2];
  acc[0] = zero_v8f(); acc[1] = zero_v8f();
  auto issue = [&](int buf, int k0) {
    async_ld_b128(xrh + (size_t)(bm * 64 + r4) * C_ + k0 + c8,
                  &sA[buf][r4 * 40 + c8]);
    async_ld_b128(kvwT + (size_t)(colBase + r4) * C_ + k0 + c8,
                  &sB[buf][r4 * 40 + c8]);
  };
  issue(0, 0);
#pragma unroll
  for (int s = 0; s < 8; ++s) {
    int buf = s & 1;
    wait_async0();
    __syncthreads();
    if (s + 1 < 8) issue(buf ^ 1, (s + 1) * 32);
    v16h a = frag_a(sA[buf] + rt * 16 * 40, 40);
    acc[0] = wmma_f16(a, frag_b(sB[buf] + ct0 * 16 * 40, 40), acc[0]);
    acc[1] = wmma_f16(a, frag_b(sB[buf] + (ct0 + 1) * 16 * 40, 40), acc[1]);
  }
  const int lane = t & 31;
  const int n = lane & 15, mh = (lane >> 4) << 3;
#pragma unroll
  for (int cc = 0; cc < 2; ++cc) {
    int c = colBase + (ct0 + cc) * 16 + n;              // 0..511
    int kvSel = c >> 8, head = (c >> 5) & 7, d = c & 31;
    _Float16* dst = kvSel ? vmat : kmat;
#pragma unroll
    for (int r = 0; r < 8; ++r) {
      int row = bm * 64 + rt * 16 + mh + r;
      int b = row >> 8, nk = row & 255;
      dst[((size_t)(b * HEADS_ + head) * NK_ + nk) * HD_ + d] = (_Float16)acc[cc][r];
    }
  }
}

// ---- attention: S=QK^T (wmma), softmax in regs, O=PV (wmma) ------------------
__global__ void __launch_bounds__(64) k_attn(const _Float16* __restrict__ qh,
                                             const _Float16* __restrict__ kmat,
                                             const _Float16* __restrict__ vmat,
                                             _Float16* __restrict__ oh) {
  __shared__ _Float16 sK[NK_ * HD_];       // [kcol][d]  16 KB
  __shared__ _Float16 sVt[HD_ * 264];      // [d][kv]    16.5 KB (padded)
  __shared__ _Float16 sP[2][16 * 264];     // per-wave P
  const int bh = blockIdx.x >> 3;          // (b,h) 0..63
  const int split = blockIdx.x & 7;
  const int t = threadIdx.x, wv = t >> 5, lane = t & 31;
  const int b = bh >> 3, h = bh & 7;
  const _Float16* Kg = kmat + (size_t)bh * NK_ * HD_;
  const _Float16* Vg = vmat + (size_t)bh * NK_ * HD_;
  for (int j = t; j < NK_ * HD_ / 8; j += 64)       // K: async DMA, contiguous
    async_ld_b128(Kg + j * 8, &sK[j * 8]);
  for (int j = t; j < NK_ * HD_; j += 64) {         // V: transposed via VGPRs
    int kv = j >> 5, d = j & 31;
    sVt[d * 264 + kv] = Vg[j];
  }
  wait_async0();
  __syncthreads();
  _Float16* sPw = sP[wv];
  const int n = lane & 15, mh = (lane >> 4) << 3;
  for (int i = 0; i < 16; ++i) {                    // 16 query blocks / wave
    int qrow0 = (split * 32 + wv * 16 + i) * 16;
    v16h qf = frag_a(qh + ((size_t)bh * N_ + qrow0) * HD_, HD_);
    v8f s[16];
#pragma unroll
    for (int tt = 0; tt < 16; ++tt)                 // S = Q K^T (K-dim = 32)
      s[tt] = wmma_f16(qf, frag_b(sK + tt * 16 * HD_, HD_), zero_v8f());
#pragma unroll
    for (int r = 0; r < 8; ++r) {                   // row softmax over 256
      float m = s[0][r];
#pragma unroll
      for (int tt = 1; tt < 16; ++tt) m = fmaxf(m, s[tt][r]);
#pragma unroll
      for (int msk = 8; msk >= 1; msk >>= 1) m = fmaxf(m, __shfl_xor(m, msk, 32));
      float sum = 0.f;
#pragma unroll
      for (int tt = 0; tt < 16; ++tt) {
        float e = __expf(s[tt][r] - m);
        s[tt][r] = e;
        sum += e;
      }
#pragma unroll
      for (int msk = 8; msk >= 1; msk >>= 1) sum += __shfl_xor(sum, msk, 32);
      float inv = 1.0f / sum;
#pragma unroll
      for (int tt = 0; tt < 16; ++tt) s[tt][r] *= inv;
    }
#pragma unroll
    for (int tt = 0; tt < 16; ++tt)                 // P -> LDS (A layout)
#pragma unroll
      for (int r = 0; r < 8; ++r)
        sPw[(mh + r) * 264 + tt * 16 + n] = (_Float16)s[tt][r];
    v8f o0 = zero_v8f(), o1 = zero_v8f();
#pragma unroll
    for (int kc = 0; kc < 8; ++kc) {                // O = P @ V
      v16h pf = frag_a(sPw + kc * 32, 264);
      o0 = wmma_f16(pf, frag_b(sVt + kc * 32, 264), o0);
      o1 = wmma_f16(pf, frag_b(sVt + 16 * 264 + kc * 32, 264), o1);
    }
#pragma unroll
    for (int r = 0; r < 8; ++r) {
      int row = qrow0 + mh + r;
      size_t base = ((size_t)b * N_ + row) * C_ + h * HD_;
      oh[base + n]      = (_Float16)o0[r];
      oh[base + 16 + n] = (_Float16)o1[r];
    }
  }
}

// ------- proj GEMM (M=32768,N=256,K=256), 256x64 tile -> f32 out + bias -------
__global__ void __launch_bounds__(256) k_gemm_proj(const _Float16* __restrict__ oh,
                                                   const _Float16* __restrict__ pwT,
                                                   const float* __restrict__ pb,
                                                   float* __restrict__ out) {
  __shared__ _Float16 sA[2][256 * 40];
  __shared__ _Float16 sB[2][64 * 40];
  const int bm = blockIdx.x;
  const int colBase = blockIdx.y * 64;
  const int t = threadIdx.x, wv = t >> 5;
  const int r4 = t >> 2, c8 = (t & 3) * 8;
  v8f acc0[4], acc1[4];
#pragma unroll
  for (int c = 0; c < 4; ++c) { acc0[c] = zero_v8f(); acc1[c] = zero_v8f(); }
  auto issue = [&](int buf, int k0) {
#pragma unroll
    for (int j = 0; j < 4; ++j)
      async_ld_b128(oh + (size_t)(bm * 256 + j * 64 + r4) * C_ + k0 + c8,
                    &sA[buf][(j * 64 + r4) * 40 + c8]);
    async_ld_b128(pwT + (size_t)(colBase + r4) * C_ + k0 + c8,
                  &sB[buf][r4 * 40 + c8]);
  };
  issue(0, 0);
#pragma unroll
  for (int s = 0; s < 8; ++s) {
    int buf = s & 1;
    wait_async0();
    __syncthreads();
    if (s + 1 < 8) issue(buf ^ 1, (s + 1) * 32);
    v16h a0 = frag_a(sA[buf] + (wv * 16) * 40, 40);
    v16h a1 = frag_a(sA[buf] + ((128 + wv * 16)) * 40, 40);
#pragma unroll
    for (int ct = 0; ct < 4; ++ct) {
      v16h bf = frag_b(sB[buf] + ct * 16 * 40, 40);
      acc0[ct] = wmma_f16(a0, bf, acc0[ct]);
      acc1[ct] = wmma_f16(a1, bf, acc1[ct]);
    }
  }
  const int lane = t & 31;
  const int n = lane & 15, mh = (lane >> 4) << 3;
#pragma unroll
  for (int half = 0; half < 2; ++half) {
    v8f* acc = half ? acc1 : acc0;
#pragma unroll
    for (int ct = 0; ct < 4; ++ct) {
      int c = colBase + ct * 16 + n;
      float bias = pb[c];
#pragma unroll
      for (int r = 0; r < 8; ++r) {
        int row = bm * 256 + half * 128 + wv * 16 + mh + r;
        out[(size_t)row * C_ + c] = acc[ct][r] + bias;
      }
    }
  }
}

extern "C" void kernel_launch(void* const* d_in, const int* in_sizes, int n_in,
                              void* d_out, int out_size, void* d_ws, size_t ws_size,
                              hipStream_t stream) {
  (void)in_sizes; (void)n_in; (void)out_size; (void)ws_size;
  const float* x   = (const float*)d_in[0];
  const float* qw  = (const float*)d_in[3];
  const float* kvw = (const float*)d_in[4];
  const float* srw = (const float*)d_in[5];
  const float* srb = (const float*)d_in[6];
  const float* lnw = (const float*)d_in[7];
  const float* lnb = (const float*)d_in[8];
  const float* pw  = (const float*)d_in[9];
  const float* pb  = (const float*)d_in[10];
  float* out = (float*)d_out;

  char* ws = (char*)d_ws;
  size_t off = 0;
  auto alloc = [&](size_t bytes) -> char* {
    char* p = ws + off;
    off += (bytes + 255) & ~(size_t)255;
    return p;
  };
  _Float16* xh   = (_Float16*)alloc((size_t)B_ * N_ * C_ * 2);          // 16 MB
  _Float16* qwT  = (_Float16*)alloc((size_t)C_ * C_ * 2);
  _Float16* kvwT = (_Float16*)alloc((size_t)C_ * 2 * C_ * 2);
  _Float16* pwT  = (_Float16*)alloc((size_t)C_ * C_ * 2);
  _Float16* srwT = (_Float16*)alloc((size_t)4096 * C_ * 2);             // 2 MB
  _Float16* qh   = (_Float16*)alloc((size_t)B_ * HEADS_ * N_ * HD_ * 2);// 16 MB
  float*    xr   = (float*)   alloc((size_t)B_ * NK_ * C_ * 4);         // 2 MB
  _Float16* xrh  = (_Float16*)alloc((size_t)B_ * NK_ * C_ * 2);         // 1 MB
  _Float16* kmat = (_Float16*)alloc((size_t)B_ * HEADS_ * NK_ * HD_ * 2);
  _Float16* vmat = (_Float16*)alloc((size_t)B_ * HEADS_ * NK_ * HD_ * 2);
  _Float16* ohh  = (_Float16*)alloc((size_t)B_ * N_ * C_ * 2);          // 16 MB

  k_pack_w<<<4096, 256, 0, stream>>>(qw, kvw, pw, srw, qwT, kvwT, pwT, srwT);
  k_cast_x<<<(B_ * N_ * C_ / 4) / 256, 256, 0, stream>>>(x, xh);
  k_gemm_q<<<dim3(128, 4), 256, 0, stream>>>(xh, qwT, qh);
  k_conv_gemm<<<dim3(32, 4), 256, 0, stream>>>(x, srwT, srb, xr);
  k_ln<<<B_ * NK_, 256, 0, stream>>>(xr, lnw, lnb, xrh);
  k_gemm_kv<<<dim3(32, 8), 256, 0, stream>>>(xrh, kvwT, kmat, vmat);
  k_attn<<<B_ * HEADS_ * 8, 64, 0, stream>>>(qh, kmat, vmat, ohh);
  k_gemm_proj<<<dim3(128, 4), 256, 0, stream>>>(ohh, pwT, pb, out);
}
